// CoGCoT_DualTower_GNN_86474871538369
// MI455X (gfx1250) — matrix operand order repaired and checked by reading, not compiled, not run on hardware
//
#include <hip/hip_runtime.h>
#include <hip/hip_bf16.h>
#include <math.h>

// ---- problem constants (match reference) ----
#define N_CKG   50000
#define E_CKG   500000
#define NP_PSG  10000
#define EP_PSG  100000
#define C_H     128
#define QD_H    768

typedef __bf16 bf16_t;
typedef __attribute__((ext_vector_type(16))) __bf16 v16bf;
typedef __attribute__((ext_vector_type(8)))  __bf16 v8bf;
typedef __attribute__((ext_vector_type(8)))  float  v8f;

__device__ __forceinline__ v16bf combine16(v8bf lo, v8bf hi) {
  v16bf r;
#pragma unroll
  for (int i = 0; i < 8; ++i) { r[i] = lo[i]; r[i + 8] = hi[i]; }
  return r;
}

// ============================================================================
// WMMA GEMM: D[M,Ncols] = A[M,128](bf16) * B[128,Ncols] + bias,  B given as
// Bt[Ncols,128] bf16 (transposed).  K fixed at 128 (4 x wmma_16x16x32_bf16).
// EPI: 0 = plain store, 1 = relu store, 2 = beta*out + (1-beta)*Xskip
// One wave per 16-row tile, loops across column tiles keeping A frags resident.
// ============================================================================
template <int EPI>
__global__ void gemm128_wmma(const bf16_t* __restrict__ A, int M,
                             const bf16_t* __restrict__ Bt, int Ncols,
                             const float* __restrict__ bias,
                             float* __restrict__ Dst,
                             const float* __restrict__ skip,
                             const float* __restrict__ Xskip) {
  const int lane  = threadIdx.x & 31;
  const int wave  = threadIdx.x >> 5;
  const int tileM = blockIdx.x * (blockDim.x >> 5) + wave;
  if (tileM * 16 >= M) return;              // wave-uniform: EXEC stays all-ones
  const int half = lane >> 4;
  const int lr   = lane & 15;

  // A fragment: lane lr holds row M=lr; element i<8 -> K=half*8+i,
  // i>=8 -> K=16+half*8+(i-8)  => two contiguous 16B chunks per K-step of 32.
  const bf16_t* Arow = A + (size_t)(tileM * 16 + lr) * 128;
  v16bf afrag[4];
#pragma unroll
  for (int kc = 0; kc < 4; ++kc) {
    v8bf lo = *(const v8bf*)(Arow + kc * 32 + half * 8);
    v8bf hi = *(const v8bf*)(Arow + kc * 32 + 16 + half * 8);
    afrag[kc] = combine16(lo, hi);
  }

  float beta = 0.0f;
  if (EPI == 2) beta = 1.0f / (1.0f + expf(-skip[0]));

  const int ntiles = Ncols >> 4;
  for (int nt = 0; nt < ntiles; ++nt) {
    // B fragment: lane holds column N=lr; element i<8 -> K=half*8+i,
    // i>=8 -> K=16+half*8+(i-8)  => two 16B chunks of Bt row.
    const bf16_t* Brow = Bt + (size_t)(nt * 16 + lr) * 128;
    v8f acc = {0.f, 0.f, 0.f, 0.f, 0.f, 0.f, 0.f, 0.f};
#pragma unroll
    for (int kc = 0; kc < 4; ++kc) {
      v8bf lo = *(const v8bf*)(Brow + kc * 32 + half * 8);
      v8bf hi = *(const v8bf*)(Brow + kc * 32 + 16 + half * 8);
      v16bf bfrag = combine16(lo, hi);
      acc = __builtin_amdgcn_wmma_f32_16x16x32_bf16(false, afrag[kc], false, bfrag,
                                                    (short)0, acc, false, false);
    }
    const int   col = nt * 16 + lr;
    const float bs  = bias[col];
#pragma unroll
    for (int r = 0; r < 8; ++r) {             // C/D layout: M = half*8 + r, N = lr
      const int row = tileM * 16 + half * 8 + r;
      float v = acc[r] + bs;
      if (EPI == 1) v = fmaxf(v, 0.0f);
      if (EPI == 2) {
        float x = Xskip[(size_t)row * Ncols + col];
        v = beta * v + (1.0f - beta) * x;
      }
      Dst[(size_t)row * Ncols + col] = v;
    }
  }
}

// ============================================================================
// Weight prep kernels (tiny)
// ============================================================================
// Fold per-head krel/vrel [H=2,64,64] into Wkqv [128,384]; emit transposed
// bf16 weight Wt [384,128] and fused f32 bias [384]. Column order: kr|q|vr.
__global__ void fuse_kqv_weights(const float* __restrict__ Wkqv,
                                 const float* __restrict__ bkqv,
                                 const float* __restrict__ krel,
                                 const float* __restrict__ vrel,
                                 bf16_t* __restrict__ Wt,
                                 float* __restrict__ bfused) {
  int idx = blockIdx.x * blockDim.x + threadIdx.x;
  if (idx >= 384 * 128) return;
  int j = idx >> 7;    // fused output column 0..383
  int c = idx & 127;   // input channel
  float val;
  if (j < 128) {
    int h = j >> 6, e = j & 63;
    const float* kr = krel + h * 4096 + e;
    const float* wr = Wkqv + (size_t)c * 384 + h * 64;
    float s = 0.f;
    for (int d = 0; d < 64; ++d) s += wr[d] * kr[(size_t)d * 64];
    val = s;
  } else if (j < 256) {
    val = Wkqv[(size_t)c * 384 + j];
  } else {
    int jj = j - 256, h = jj >> 6, e = jj & 63;
    const float* vr = vrel + h * 4096 + e;
    const float* wr = Wkqv + (size_t)c * 384 + 256 + h * 64;
    float s = 0.f;
    for (int d = 0; d < 64; ++d) s += wr[d] * vr[(size_t)d * 64];
    val = s;
  }
  Wt[(size_t)j * 128 + c] = (bf16_t)val;
  if (c == 0) {
    float bv;
    if (j < 128) {
      int h = j >> 6, e = j & 63;
      float s = 0.f;
      for (int d = 0; d < 64; ++d) s += bkqv[h * 64 + d] * krel[h * 4096 + d * 64 + e];
      bv = s;
    } else if (j < 256) {
      bv = bkqv[j];
    } else {
      int jj = j - 256, h = jj >> 6, e = jj & 63;
      float s = 0.f;
      for (int d = 0; d < 64; ++d) s += bkqv[256 + h * 64 + d] * vrel[h * 4096 + d * 64 + e];
      bv = s;
    }
    bfused[j] = bv;
  }
}

// W [128,128] f32 -> Wt [128,128] bf16 transposed
__global__ void transpose128_to_bf16(const float* __restrict__ W,
                                     bf16_t* __restrict__ Wt) {
  int idx = blockIdx.x * blockDim.x + threadIdx.x;   // 16384 threads
  if (idx >= 128 * 128) return;
  int j = idx >> 7, c = idx & 127;
  Wt[idx] = (bf16_t)W[(size_t)c * 128 + j];
}

// ============================================================================
// Elementwise kernels
// ============================================================================
__global__ void cvt_to_bf16(const float* __restrict__ x, bf16_t* __restrict__ y, size_t n) {
  size_t i = (size_t)blockIdx.x * blockDim.x + threadIdx.x;
  if (i < n) y[i] = (bf16_t)x[i];
}

__global__ void fill_f32(float* __restrict__ p, float v, size_t n) {
  size_t i = (size_t)blockIdx.x * blockDim.x + threadIdx.x;
  if (i < n) p[i] = v;
}

// x += fused (broadcast over rows, C=128); emit bf16 copy of result
__global__ void add_fused_cvt(float* __restrict__ x, const float* __restrict__ fused,
                              bf16_t* __restrict__ xb, size_t n) {
  size_t i = (size_t)blockIdx.x * blockDim.x + threadIdx.x;
  if (i >= n) return;
  float v = x[i] + fused[i & 127];
  x[i]  = v;
  xb[i] = (bf16_t)v;
}

// exact GELU then bf16 (feeds the Wout GEMM)
__global__ void gelu_cvt(const float* __restrict__ a, bf16_t* __restrict__ b, size_t n) {
  size_t i = (size_t)blockIdx.x * blockDim.x + threadIdx.x;
  if (i >= n) return;
  float v = a[i];
  float g = 0.5f * v * (1.0f + erff(v * 0.70710678118654752f));
  b[i] = (bf16_t)g;
}

// qv = relu(query_emb[768] @ W[768,128] + b)   (tiny; one block of 128)
__global__ void query_proj(const float* __restrict__ qe, const float* __restrict__ W,
                           const float* __restrict__ b, float* __restrict__ qv) {
  int c = threadIdx.x;
  float s = b[c];
  for (int k = 0; k < QD_H; ++k) s += qe[k] * W[(size_t)k * 128 + c];
  qv[c] = fmaxf(s, 0.0f);
}

// ============================================================================
// Edge-phase kernels (L2-resident gathers; wave per edge)
// ============================================================================
// kqv rows are [kr(128) | q(128) | vr(128)], stride 384 floats.
__global__ void edge_score(const float* __restrict__ kqv_src,
                           const float* __restrict__ kqv_dst,
                           const int* __restrict__ src, const int* __restrict__ dst,
                           const float* __restrict__ prel,
                           float* __restrict__ sc, int E) {
  int e = blockIdx.x * (blockDim.x >> 5) + (threadIdx.x >> 5);
  if (e >= E) return;
  int lane = threadIdx.x & 31;
  int si = src[e], di = dst[e];
  float4 kv = ((const float4*)(kqv_src + (size_t)si * 384))[lane];
  float4 qv = ((const float4*)(kqv_dst + (size_t)di * 384 + 128))[lane];
  float p = kv.x * qv.x + kv.y * qv.y + kv.z * qv.z + kv.w * qv.w;
  // lanes 0..15 cover head0 dims, 16..31 head1 -> reduce within half-wave
  p += __shfl_xor(p, 1, 32);
  p += __shfl_xor(p, 2, 32);
  p += __shfl_xor(p, 4, 32);
  p += __shfl_xor(p, 8, 32);
  if ((lane & 15) == 0) {
    int h = lane >> 4;
    sc[(size_t)e * 2 + h] = p * prel[h] * 0.125f;   // 1/sqrt(D=64)
  }
}

__device__ __forceinline__ unsigned f32_ord(float x) {
  unsigned u = __float_as_uint(x);
  return (u & 0x80000000u) ? ~u : (u | 0x80000000u);
}
__device__ __forceinline__ float ord_f32(unsigned u) {
  unsigned b = (u & 0x80000000u) ? (u & 0x7FFFFFFFu) : ~u;
  return __uint_as_float(b);
}

__global__ void seg_max(const float* __restrict__ sc, const int* __restrict__ dst,
                        unsigned* __restrict__ m_enc, int E) {
  int i = blockIdx.x * blockDim.x + threadIdx.x;
  if (i >= E * 2) return;
  int e = i >> 1, h = i & 1;
  atomicMax(&m_enc[(size_t)dst[e] * 2 + h], f32_ord(sc[i]));
}

__global__ void exp_sum(float* __restrict__ sc, const int* __restrict__ dst,
                        const unsigned* __restrict__ m_enc, float* __restrict__ ssum, int E) {
  int i = blockIdx.x * blockDim.x + threadIdx.x;
  if (i >= E * 2) return;
  int e = i >> 1, h = i & 1;
  size_t seg = (size_t)dst[e] * 2 + h;
  float m  = ord_f32(m_enc[seg]);
  float ev = expf(sc[i] - m);
  sc[i] = ev;
  atomicAdd(&ssum[seg], ev);
}

__global__ void edge_aggregate(const float* __restrict__ sc, const int* __restrict__ src,
                               const int* __restrict__ dst, const float* __restrict__ ssum,
                               const float* __restrict__ kqv_src, float* __restrict__ agg,
                               int E) {
  int e = blockIdx.x * (blockDim.x >> 5) + (threadIdx.x >> 5);
  if (e >= E) return;
  int lane = threadIdx.x & 31;
  int h = lane >> 4;                      // cols lane*4..lane*4+3 all in head lane/16
  int si = src[e], di = dst[e];
  float a = sc[(size_t)e * 2 + h] / (ssum[(size_t)di * 2 + h] + 1e-16f);
  float4 vv = ((const float4*)(kqv_src + (size_t)si * 384 + 256))[lane];
  float* dp = agg + (size_t)di * 128 + lane * 4;
  atomicAdd(dp + 0, a * vv.x);
  atomicAdd(dp + 1, a * vv.y);
  atomicAdd(dp + 2, a * vv.z);
  atomicAdd(dp + 3, a * vv.w);
}

// ============================================================================
// Reductions (column sums for mean pools)
// ============================================================================
__global__ void colsum(const float* __restrict__ X, int nrows, float* __restrict__ out) {
  int c  = threadIdx.x;                 // 0..127
  int r0 = blockIdx.x * 256;
  int r1 = r0 + 256; if (r1 > nrows) r1 = nrows;
  float s = 0.f;
  for (int r = r0; r < r1; ++r) s += X[(size_t)r * 128 + c];
  atomicAdd(&out[c], s);
}

__global__ void make_fused(const float* __restrict__ qv, const float* __restrict__ csum,
                           float inv, float* __restrict__ fused) {
  int c = threadIdx.x;
  fused[c] = qv[c] + csum[c] * inv;
}

__global__ void write_thought(const float* __restrict__ csum, float inv,
                              float* __restrict__ out) {
  int c = threadIdx.x;
  out[c] = csum[c] * inv;
}

// ============================================================================
// Host orchestration
// ============================================================================
extern "C" void kernel_launch(void* const* d_in, const int* in_sizes, int n_in,
                              void* d_out, int out_size, void* d_ws, size_t ws_size,
                              hipStream_t stream) {
  (void)in_sizes; (void)n_in; (void)out_size; (void)ws_size;

  const float* ckg_x_dis  = (const float*)d_in[0];
  const float* ckg_x_drug = (const float*)d_in[1];
  const float* psg_x      = (const float*)d_in[2];
  const float* query_emb  = (const float*)d_in[3];
  const float* ckg_lin_W  = (const float*)d_in[4];
  const float* ckg_lin_b  = (const float*)d_in[5];
  const float* psg_lin_W  = (const float*)d_in[6];
  const float* psg_lin_b  = (const float*)d_in[7];
  const float* query_W    = (const float*)d_in[8];
  const float* query_b    = (const float*)d_in[9];
  const float* ckg_Wkqv   = (const float*)d_in[10];
  const float* ckg_bkqv   = (const float*)d_in[11];
  const float* ckg_Wout   = (const float*)d_in[12];
  const float* ckg_bout   = (const float*)d_in[13];
  const float* ckg_skip   = (const float*)d_in[14];
  const float* ckg_krel   = (const float*)d_in[15];
  const float* ckg_vrel   = (const float*)d_in[16];
  const float* ckg_prel   = (const float*)d_in[17];
  const float* psg_Wkqv   = (const float*)d_in[18];
  const float* psg_bkqv   = (const float*)d_in[19];
  const float* psg_Wout   = (const float*)d_in[20];
  const float* psg_bout   = (const float*)d_in[21];
  const float* psg_skip   = (const float*)d_in[22];
  const float* psg_krel   = (const float*)d_in[23];
  const float* psg_vrel   = (const float*)d_in[24];
  const float* psg_prel   = (const float*)d_in[25];
  const int* ei_src[2] = { (const int*)d_in[26], (const int*)d_in[28] };
  const int* ei_dst[2] = { (const int*)d_in[27], (const int*)d_in[29] };
  const int* psg_src = (const int*)d_in[30];
  const int* psg_dst = (const int*)d_in[31];

  const size_t NC  = (size_t)N_CKG * C_H;     // 6.4M
  const size_t NPC = (size_t)NP_PSG * C_H;

  // ---- carve workspace (256B aligned) ----
  char* base = (char*)d_ws;
  size_t off = 0;
  auto carve = [&](size_t bytes) -> void* {
    void* p = base + off;
    off = (off + bytes + 255) & ~(size_t)255;
    return p;
  };
  float*  xA[2]   = { (float*)carve(NC * 4), (float*)carve(NC * 4) };
  bf16_t* xb[2]   = { (bf16_t*)carve(NC * 2), (bf16_t*)carve(NC * 2) };
  float*  kqvr[2] = { (float*)carve(NC * 3 * 4), (float*)carve(NC * 3 * 4) };
  float*  agg[2]  = { (float*)carve(NC * 4), (float*)carve(NC * 4) };
  bf16_t* aggb    = (bf16_t*)carve(NC * 2);
  float*  sc      = (float*)carve((size_t)E_CKG * 2 * 4);
  unsigned* m_enc = (unsigned*)carve((size_t)N_CKG * 2 * 4);
  float*  ssum    = (float*)carve((size_t)N_CKG * 2 * 4);
  float*  pxA     = (float*)carve(NPC * 4);
  bf16_t* pxb     = (bf16_t*)carve(NPC * 2);
  float*  pxnew   = (float*)carve(NPC * 4);
  float*  kqvp    = (float*)carve(NPC * 3 * 4);
  float*  aggp    = (float*)carve(NPC * 4);
  bf16_t* linT0   = (bf16_t*)carve(16384 * 2);
  bf16_t* linT1   = (bf16_t*)carve(16384 * 2);
  bf16_t* psglinT = (bf16_t*)carve(16384 * 2);
  bf16_t* WkqvT[2]= { (bf16_t*)carve(49152 * 2), (bf16_t*)carve(49152 * 2) };
  float*  bkf[2]  = { (float*)carve(384 * 4), (float*)carve(384 * 4) };
  bf16_t* psgWT   = (bf16_t*)carve(49152 * 2);
  float*  psgbf   = (float*)carve(384 * 4);
  bf16_t* WoutT   = (bf16_t*)carve(16384 * 2);
  float*  qvv     = (float*)carve(128 * 4);
  float*  fused   = (float*)carve(128 * 4);
  float*  csum    = (float*)carve(128 * 4);

  auto eg  = [](size_t n) { return dim3((unsigned)((n + 255) / 256)); };   // elementwise
  auto gg  = [](int M)    { return dim3((unsigned)(((M + 15) / 16 + 7) / 8)); }; // gemm (8 waves/blk)
  auto wg  = [](int E)    { return dim3((unsigned)((E + 7) / 8)); };       // wave-per-edge

  // ================= prep =================
  cvt_to_bf16<<<eg(NC), 256, 0, stream>>>(ckg_x_dis,  xb[0], NC);
  cvt_to_bf16<<<eg(NC), 256, 0, stream>>>(ckg_x_drug, xb[1], NC);
  cvt_to_bf16<<<eg(NPC), 256, 0, stream>>>(psg_x, pxb, NPC);
  transpose128_to_bf16<<<64, 256, 0, stream>>>(ckg_lin_W,         linT0);
  transpose128_to_bf16<<<64, 256, 0, stream>>>(ckg_lin_W + 16384, linT1);
  transpose128_to_bf16<<<64, 256, 0, stream>>>(psg_lin_W,         psglinT);
  gemm128_wmma<1><<<gg(N_CKG), 256, 0, stream>>>(xb[0], N_CKG, linT0, 128, ckg_lin_b,       xA[0], nullptr, nullptr);
  gemm128_wmma<1><<<gg(N_CKG), 256, 0, stream>>>(xb[1], N_CKG, linT1, 128, ckg_lin_b + 128, xA[1], nullptr, nullptr);
  gemm128_wmma<1><<<gg(NP_PSG), 256, 0, stream>>>(pxb, NP_PSG, psglinT, 128, psg_lin_b,     pxA,   nullptr, nullptr);
  query_proj<<<1, 128, 0, stream>>>(query_emb, query_W, query_b, qvv);

  // ================= PSG tower (one HGT conv + mean pool) =================
  fuse_kqv_weights<<<192, 256, 0, stream>>>(psg_Wkqv, psg_bkqv, psg_krel, psg_vrel, psgWT, psgbf);
  cvt_to_bf16<<<eg(NPC), 256, 0, stream>>>(pxA, pxb, NPC);
  gemm128_wmma<0><<<gg(NP_PSG), 256, 0, stream>>>(pxb, NP_PSG, psgWT, 384, psgbf, kqvp, nullptr, nullptr);
  fill_f32<<<eg(NP_PSG * 2), 256, 0, stream>>>((float*)m_enc, 0.0f, (size_t)NP_PSG * 2);
  fill_f32<<<eg(NP_PSG * 2), 256, 0, stream>>>(ssum, 0.0f, (size_t)NP_PSG * 2);
  fill_f32<<<eg(NPC), 256, 0, stream>>>(aggp, 0.0f, NPC);
  edge_score<<<wg(EP_PSG), 256, 0, stream>>>(kqvp, kqvp, psg_src, psg_dst, psg_prel, sc, EP_PSG);
  seg_max<<<eg((size_t)EP_PSG * 2), 256, 0, stream>>>(sc, psg_dst, m_enc, EP_PSG);
  exp_sum<<<eg((size_t)EP_PSG * 2), 256, 0, stream>>>(sc, psg_dst, m_enc, ssum, EP_PSG);
  edge_aggregate<<<wg(EP_PSG), 256, 0, stream>>>(sc, psg_src, psg_dst, ssum, kqvp, aggp, EP_PSG);
  gelu_cvt<<<eg(NPC), 256, 0, stream>>>(aggp, aggb, NPC);
  transpose128_to_bf16<<<64, 256, 0, stream>>>(psg_Wout, WoutT);
  gemm128_wmma<2><<<gg(NP_PSG), 256, 0, stream>>>(aggb, NP_PSG, WoutT, 128, psg_bout, pxnew, psg_skip, pxA);
  fill_f32<<<1, 128, 0, stream>>>(csum, 0.0f, 128);
  colsum<<<(NP_PSG + 255) / 256, 128, 0, stream>>>(pxnew, NP_PSG, csum);
  make_fused<<<1, 128, 0, stream>>>(qvv, csum, 1.0f / (float)NP_PSG, fused);

  // ================= CKG think steps =================
  for (int i = 0; i < 2; ++i) {
    for (int t = 0; t < 2; ++t) {
      add_fused_cvt<<<eg(NC), 256, 0, stream>>>(xA[t], fused, xb[t], NC);
      int s = i * 2 + t;
      fuse_kqv_weights<<<192, 256, 0, stream>>>(ckg_Wkqv + (size_t)s * 49152,
                                                ckg_bkqv + (size_t)s * 384,
                                                ckg_krel + (size_t)s * 8192,
                                                ckg_vrel + (size_t)s * 8192,
                                                WkqvT[t], bkf[t]);
      gemm128_wmma<0><<<gg(N_CKG), 256, 0, stream>>>(xb[t], N_CKG, WkqvT[t], 384, bkf[t],
                                                     kqvr[t], nullptr, nullptr);
    }
    // relations: r=0 (dis->drug), r=1 (drug->dis); dst type = 1-r
    for (int r = 0; r < 2; ++r) {
      int st = r, dt = 1 - r;
      fill_f32<<<eg((size_t)N_CKG * 2), 256, 0, stream>>>((float*)m_enc, 0.0f, (size_t)N_CKG * 2);
      fill_f32<<<eg((size_t)N_CKG * 2), 256, 0, stream>>>(ssum, 0.0f, (size_t)N_CKG * 2);
      fill_f32<<<eg(NC), 256, 0, stream>>>(agg[dt], 0.0f, NC);
      edge_score<<<wg(E_CKG), 256, 0, stream>>>(kqvr[st], kqvr[dt], ei_src[r], ei_dst[r],
                                                ckg_prel + (size_t)(i * 2 + r) * 2, sc, E_CKG);
      seg_max<<<eg((size_t)E_CKG * 2), 256, 0, stream>>>(sc, ei_dst[r], m_enc, E_CKG);
      exp_sum<<<eg((size_t)E_CKG * 2), 256, 0, stream>>>(sc, ei_dst[r], m_enc, ssum, E_CKG);
      edge_aggregate<<<wg(E_CKG), 256, 0, stream>>>(sc, ei_src[r], ei_dst[r], ssum,
                                                    kqvr[st], agg[dt], E_CKG);
    }
    for (int t = 0; t < 2; ++t) {
      int s = i * 2 + t;
      gelu_cvt<<<eg(NC), 256, 0, stream>>>(agg[t], aggb, NC);
      transpose128_to_bf16<<<64, 256, 0, stream>>>(ckg_Wout + (size_t)s * 16384, WoutT);
      float* dstp = (i == 0) ? xA[t] : ((float*)d_out + (size_t)t * NC);
      gemm128_wmma<2><<<gg(N_CKG), 256, 0, stream>>>(aggb, N_CKG, WoutT, 128,
                                                     ckg_bout + (size_t)s * 128, dstp,
                                                     ckg_skip + s, xA[t]);
    }
    fill_f32<<<1, 128, 0, stream>>>(csum, 0.0f, 128);
    const float* t0 = (i == 0) ? xA[0] : (const float*)d_out;
    const float* t1 = (i == 0) ? xA[1] : ((const float*)d_out + NC);
    colsum<<<(N_CKG + 255) / 256, 128, 0, stream>>>(t0, N_CKG, csum);
    colsum<<<(N_CKG + 255) / 256, 128, 0, stream>>>(t1, N_CKG, csum);
    write_thought<<<1, 128, 0, stream>>>(csum, 1.0f / (2.0f * (float)N_CKG),
                                         (float*)d_out + 2 * NC + (size_t)i * 128);
  }
}
